// Head_63178968924731
// MI455X (gfx1250) — 16-bit and 8-bit matrix operand order repaired and checked by reading, compiled, not run on hardware
//
#include <hip/hip_runtime.h>

typedef __attribute__((ext_vector_type(16))) int    v16i;
typedef __attribute__((ext_vector_type(8)))  float  v8f;
typedef __attribute__((ext_vector_type(16))) __bf16 v16bf;

#define B_SZ 512
#define T_SZ 256
#define C_SZ 96
#define H_SZ 16
#define NW   8

#if __has_builtin(__builtin_amdgcn_cvt_pk_bf8_f32) && \
    __has_builtin(__builtin_amdgcn_cvt_f32_bf8)
#define HW_BF8 1
#else
#define HW_BF8 0
#endif

struct U32x8 { uint4 lo, hi; };

__device__ __forceinline__ unsigned rnd_bf8(float x) {
  const unsigned short u = __builtin_bit_cast(unsigned short, (_Float16)x);
  return (unsigned)u + 0x7Fu + ((u >> 8) & 1u);
}
__device__ __forceinline__ int pack_bf8x4(float4 f) {
#if HW_BF8
  int p = __builtin_amdgcn_cvt_pk_bf8_f32(f.x, f.y, 0, false);
  return __builtin_amdgcn_cvt_pk_bf8_f32(f.z, f.w, p, true);
#else
  const unsigned lo = rnd_bf8(f.x) | (rnd_bf8(f.y) << 16);
  const unsigned hi = rnd_bf8(f.z) | (rnd_bf8(f.w) << 16);
  return (int)__builtin_amdgcn_perm(hi, lo, 0x07050301u);
#endif
}
__device__ __forceinline__ float bf8val(float x) {
#if HW_BF8
  return __builtin_amdgcn_cvt_f32_bf8(
      __builtin_amdgcn_cvt_pk_bf8_f32(x, 0.0f, 0, false), 0);
#else
  const unsigned short b = (unsigned short)(rnd_bf8(x) & 0xFF00u);
  return (float)__builtin_bit_cast(_Float16, b);
#endif
}
__device__ __forceinline__ unsigned bf8byte(float x) {
#if HW_BF8
  return (unsigned)__builtin_amdgcn_cvt_pk_bf8_f32(x, 0.0f, 0, false) & 0xFFu;
#else
  return (rnd_bf8(x) >> 8) & 0xFFu;
#endif
}
template <int SEL>
__device__ __forceinline__ float bf8_decode(unsigned dw) {
#if HW_BF8
  return __builtin_amdgcn_cvt_f32_bf8((int)dw, SEL);
#else
  return (float)__builtin_bit_cast(
      _Float16, (unsigned short)(((dw >> (8 * SEL)) & 0xFFu) << 8));
#endif
}

__global__ __launch_bounds__(256) void qattn_head_kernel(
    const float* __restrict__ x,  const float* __restrict__ Wk,
    const float* __restrict__ Wq, const float* __restrict__ Wv,
    float* __restrict__ out)
{
  __shared__ __align__(16) __bf16 q_lds[T_SZ][H_SZ];
  __shared__ __align__(16) __bf16 k_lds[T_SZ][H_SZ];
  __shared__ __align__(16) __bf16 vT_lds[H_SZ][T_SZ];
  __shared__ __align__(16) __bf16 zpad[16];
  __shared__ __align__(16) unsigned char wei_lds[NW][16][T_SZ];

  const int b    = blockIdx.x;
  const int tid  = threadIdx.x;
  const int wv   = tid >> 5;
  const int lane = tid & 31;
  const int hl   = lane >> 4;
  const int l16  = lane & 15;

  if (tid < 16) zpad[tid] = (__bf16)0.0f;

  const float* xb = x + (size_t)b * T_SZ * C_SZ;
  const float* Ws[3] = {Wq, Wk, Wv};

  v16i Bw[3];
#pragma unroll
  for (int m = 0; m < 3; ++m) {
    const float* Wr = Ws[m] + l16 * C_SZ;
#pragma unroll
    for (int i = 0; i < 16; ++i) {
      const int k0 = ((i >> 3) * 64) + (((i & 7) >> 1) * 16) + hl * 8 + ((i & 1) * 4);
      Bw[m][i] = (k0 < C_SZ) ? pack_bf8x4(*(const float4*)(Wr + k0)) : 0;
    }
  }

#pragma unroll 1
  for (int t = wv; t < 16; t += NW) {
    const int    row = t * 16 + l16;
    const float* xr  = xb + row * C_SZ;

    v16i A;
#pragma unroll
    for (int i = 0; i < 16; ++i) {
      const int k0 = ((i >> 3) * 64) + (((i & 7) >> 1) * 16) + hl * 8 + ((i & 1) * 4);
      A[i] = (k0 < C_SZ) ? pack_bf8x4(*(const float4*)(xr + k0)) : 0;
    }

#pragma unroll
    for (int m = 0; m < 3; ++m) {
      v8f acc = {};
      acc = __builtin_amdgcn_wmma_f32_16x16x128_bf8_bf8(A, Bw[m], (short)0, acc,
                                                        false, false);
#pragma unroll
      for (int r = 0; r < 8; ++r) {
        const int mrow = t * 16 + hl * 8 + r;
        const __bf16 hb = (__bf16)acc[r];
        if (m == 0)      q_lds[mrow][l16]  = hb;
        else if (m == 1) k_lds[mrow][l16]  = hb;
        else             vT_lds[l16][mrow] = hb;
      }
    }
  }
  __syncthreads();

  const float scale = 0.25f;
#pragma unroll 1
  for (int t = wv; t < 16; t += NW) {
    v16bf Aq;
    {
      const __bf16* qp = &q_lds[t * 16 + l16][hl * 8];
      U32x8 u; u.lo = *(const uint4*)qp; u.hi = uint4{0, 0, 0, 0};
      Aq = __builtin_bit_cast(v16bf, u);
    }

    v8f S[16];
#pragma unroll
    for (int s = 0; s < 16; ++s) {
      if (s <= t) {
        const __bf16* kp = &k_lds[s * 16 + l16][hl * 8];
        U32x8 u; u.lo = *(const uint4*)kp; u.hi = uint4{0, 0, 0, 0};
        const v16bf Bk = __builtin_bit_cast(v16bf, u);
        v8f z = {};
        v8f sc = __builtin_amdgcn_wmma_f32_16x16x32_bf16(false, Aq, false, Bk,
                                                         (short)0, z, false, false);
#pragma unroll
        for (int r = 0; r < 8; ++r) {
          const bool masked = (s == t) && (l16 > (hl * 8 + r));
          S[s][r] = masked ? -INFINITY : sc[r] * scale;
        }
      } else {
#pragma unroll
        for (int r = 0; r < 8; ++r) S[s][r] = -INFINITY;
      }
    }

    float rmax[8], rinv[8];
#pragma unroll
    for (int r = 0; r < 8; ++r) {
      float m = -INFINITY;
#pragma unroll
      for (int s = 0; s < 16; ++s) m = fmaxf(m, S[s][r]);
#pragma unroll
      for (int d = 1; d < 16; d <<= 1) m = fmaxf(m, __shfl_xor(m, d, 16));
      rmax[r] = m;
    }
#pragma unroll
    for (int r = 0; r < 8; ++r) {
      float sum = 0.0f;
#pragma unroll
      for (int s = 0; s < 16; ++s) {
        if (s <= t) {
          const float pq = bf8val(__expf(S[s][r] - rmax[r]));
          S[s][r] = pq;
          sum += pq;
        }
      }
#pragma unroll
      for (int d = 1; d < 16; d <<= 1) sum += __shfl_xor(sum, d, 16);
      rinv[r] = 1.0f / bf8val(sum);
    }

#pragma unroll
    for (int s = 0; s < 16; ++s) {
#pragma unroll
      for (int r = 0; r < 8; ++r) {
        unsigned wb = 0;
        if (s <= t) wb = bf8byte(S[s][r] * rinv[r]);
        wei_lds[wv][hl * 8 + r][s * 16 + l16] = (unsigned char)wb;
      }
    }

    v8f O = {};
    const int cmax = (t * 16 + 15) >> 5;
#pragma unroll
    for (int c = 0; c < 8; ++c) {
      if (c <= cmax) {
        v16bf Aw, Bv;
        const unsigned* wp =
            (const unsigned*)&wei_lds[wv][l16][c * 32 + hl * 8];
        const unsigned w0 = wp[0], w1 = wp[1];
        const unsigned w2 = wp[4], w3 = wp[5];
        Aw[0]  = (__bf16)bf8_decode<0>(w0);
        Aw[1]  = (__bf16)bf8_decode<1>(w0);
        Aw[2]  = (__bf16)bf8_decode<2>(w0);
        Aw[3]  = (__bf16)bf8_decode<3>(w0);
        Aw[4]  = (__bf16)bf8_decode<0>(w1);
        Aw[5]  = (__bf16)bf8_decode<1>(w1);
        Aw[6]  = (__bf16)bf8_decode<2>(w1);
        Aw[7]  = (__bf16)bf8_decode<3>(w1);
        Aw[8]  = (__bf16)bf8_decode<0>(w2);
        Aw[9]  = (__bf16)bf8_decode<1>(w2);
        Aw[10] = (__bf16)bf8_decode<2>(w2);
        Aw[11] = (__bf16)bf8_decode<3>(w2);
        Aw[12] = (__bf16)bf8_decode<0>(w3);
        Aw[13] = (__bf16)bf8_decode<1>(w3);
        Aw[14] = (__bf16)bf8_decode<2>(w3);
        Aw[15] = (__bf16)bf8_decode<3>(w3);
        {
          const __bf16* vp = &vT_lds[l16][c * 32 + hl * 8];
          U32x8 u; u.lo = *(const uint4*)vp; u.hi = *(const uint4*)(vp + 16);
          Bv = __builtin_bit_cast(v16bf, u);
        }
        O = __builtin_amdgcn_wmma_f32_16x16x32_bf16(false, Aw, false, Bv,
                                                    (short)0, O, false, false);
      }
    }

    float* op = out + (((size_t)b * T_SZ) + (size_t)t * 16 + hl * 8) * H_SZ + l16;
#pragma unroll
    for (int r = 0; r < 8; ++r)
      op[(size_t)r * H_SZ] = O[r];
  }
}

extern "C" void kernel_launch(void* const* d_in, const int* in_sizes, int n_in,
                              void* d_out, int out_size, void* d_ws, size_t ws_size,
                              hipStream_t stream) {
  const float* x  = (const float*)d_in[0];
  const float* Wk = (const float*)d_in[1];
  const float* Wq = (const float*)d_in[2];
  const float* Wv = (const float*)d_in[3];
  float* out = (float*)d_out;
  qattn_head_kernel<<<dim3(B_SZ), dim3(256), 0, stream>>>(x, Wk, Wq, Wv, out);
}
